// HungarianMatcher_57208964382737
// MI455X (gfx1250) — matrix operand order repaired and checked by reading, not compiled, not run on hardware
//
#include <hip/hip_runtime.h>
#include <hip/hip_bf16.h>

// ---------------------------------------------------------------------------
// Problem constants (from the reference): bs=16, nq=900, d=256, M=1600
// ---------------------------------------------------------------------------
#define NROWS 14400   // bs*nq
#define NDIM  256
#define NTGT  1600

typedef _Float16 v16h __attribute__((ext_vector_type(16)));
typedef _Float16 v8h  __attribute__((ext_vector_type(8)));
typedef float    v8f  __attribute__((ext_vector_type(8)));
typedef float    v4f  __attribute__((ext_vector_type(4)));

// ---------------------------------------------------------------------------
// Kernel 1: row-normalize pred_logits -> f16 (A), convert tgt_embs -> f16 (B)
//   blocks [0, NROWS)            : normalize + convert one logits row
//   blocks [NROWS, NROWS+NTGT)   : plain convert one tgt_embs row
// ---------------------------------------------------------------------------
__global__ __launch_bounds__(256)
void matcher_prep_kernel(const float* __restrict__ logits,
                         const float* __restrict__ embs,
                         _Float16* __restrict__ A,
                         _Float16* __restrict__ B)
{
    const int row = blockIdx.x;
    const int tid = threadIdx.x;   // 0..255, one element of the row

    if (row < NROWS) {
        float x = logits[row * NDIM + tid];
        float s = x * x;
        // wave32 tree reduction
        #pragma unroll
        for (int off = 16; off > 0; off >>= 1)
            s += __shfl_xor(s, off, 32);
        __shared__ float part[8];
        if ((tid & 31) == 0) part[tid >> 5] = s;
        __syncthreads();
        float tot = part[0] + part[1] + part[2] + part[3]
                  + part[4] + part[5] + part[6] + part[7];
        float inv = rsqrtf(tot);
        A[row * NDIM + tid] = (_Float16)(x * inv);
    } else {
        const int r = row - NROWS;
        B[r * NDIM + tid] = (_Float16)embs[r * NDIM + tid];
    }
}

// ---------------------------------------------------------------------------
// Kernel 2: fused cost volume.
//   One 16x16 output tile per wave32. Block = 4 waves = 4 column tiles.
//   grid = (NROWS/16, (NTGT/16)/4) = (900, 25)
//
//   cost_class via v_wmma_f32_16x16x32_f16 over K=256 (8 steps),
//   then per-element L1 bbox + GIoU + sigmoid in the epilogue.
// ---------------------------------------------------------------------------
__global__ __launch_bounds__(128)
void matcher_cost_kernel(const _Float16* __restrict__ A,   // [NROWS, 256] normalized f16
                         const _Float16* __restrict__ B,   // [NTGT, 256] f16
                         const float*    __restrict__ pred_boxes, // [NROWS,4] cxcywh
                         const float*    __restrict__ tgt_bbox,   // [NTGT,4]  cxcywh
                         float*          __restrict__ out)        // [NROWS, NTGT]
{
    const int lane = threadIdx.x & 31;
    const int wave = threadIdx.x >> 5;
    const int g    = lane >> 4;     // half-wave group (0/1)
    const int l16  = lane & 15;

    const int row_base = blockIdx.x * 16;
    const int col_base = (blockIdx.y * 4 + wave) * 16;

    // A-fragment source: row (row_base + l16); 16-bit A 16x32 layout:
    //   lanes 0-15 hold K = {0..7, 16..23}, lanes 16-31 hold K = {8..15, 24..31}
    const _Float16* __restrict__ arow = A + (size_t)(row_base + l16) * NDIM;
    // B-fragment source: column n = target row (col_base + l16); 16-bit B 32x16:
    //   lanes 0-15 hold K = {0..7, 16..23}, lanes 16-31 hold K = {8..15, 24..31}
    const _Float16* __restrict__ brow = B + (size_t)(col_base + l16) * NDIM;

    v8f acc = {};   // D/C layout: VGPR j -> M = j + 8*g, N = l16

    #pragma unroll
    for (int kk = 0; kk < NDIM; kk += 32) {
        union { v16h v; v8h h[2]; } a;
        a.h[0] = *(const v8h*)(arow + kk +      8 * g);
        a.h[1] = *(const v8h*)(arow + kk + 16 + 8 * g);
        union { v16h v; v8h h[2]; } b;
        b.h[0] = *(const v8h*)(brow + kk +      8 * g);
        b.h[1] = *(const v8h*)(brow + kk + 16 + 8 * g);
        v16h bv = b.v;
        acc = __builtin_amdgcn_wmma_f32_16x16x32_f16(
                  /*neg_a=*/false, a.v, /*neg_b=*/false, bv,
                  /*c_mod=*/(short)0, acc, /*reuse_a=*/false, /*reuse_b=*/false);
    }

    // ---- epilogue: box costs + sigmoid, fused with the store ----
    // Target box is constant per lane.
    v4f t = *(const v4f*)(tgt_bbox + (size_t)(col_base + l16) * 4);
    const float tx1 = t.x - 0.5f * t.z, ty1 = t.y - 0.5f * t.w;
    const float tx2 = t.x + 0.5f * t.z, ty2 = t.y + 0.5f * t.w;
    const float area_t = t.z * t.w;

    const float* __restrict__ pb = pred_boxes + (size_t)(row_base + 8 * g) * 4;

    #pragma unroll
    for (int j = 0; j < 8; ++j) {
        v4f p = *(const v4f*)(pb + j * 4);

        float cls = fmaxf(acc[j], 0.0f);                       // relu(cost_class)

        float l1 = fabsf(p.x - t.x) + fabsf(p.y - t.y)
                 + fabsf(p.z - t.z) + fabsf(p.w - t.w);        // cost_bbox

        float px1 = p.x - 0.5f * p.z, py1 = p.y - 0.5f * p.w;
        float px2 = p.x + 0.5f * p.z, py2 = p.y + 0.5f * p.w;
        float area_p = p.z * p.w;

        float iw = fminf(px2, tx2) - fmaxf(px1, tx1);
        float ih = fminf(py2, ty2) - fmaxf(py1, ty1);
        float inter = fmaxf(iw, 0.0f) * fmaxf(ih, 0.0f);
        float uni   = area_p + area_t - inter;
        float iou   = inter / uni;

        float ew = fmaxf(px2, tx2) - fminf(px1, tx1);
        float eh = fmaxf(py2, ty2) - fminf(py1, ty1);
        float earea = fmaxf(ew, 0.0f) * fmaxf(eh, 0.0f);
        float giou  = iou - (earea - uni) / earea;

        float C = l1 + cls - giou;                             // cost_giou = -giou
        float sig = 1.0f / (1.0f + __expf(-C));

        out[(size_t)(row_base + 8 * g + j) * NTGT + col_base + l16] = sig;
    }
}

// ---------------------------------------------------------------------------
// Launch
// ---------------------------------------------------------------------------
extern "C" void kernel_launch(void* const* d_in, const int* in_sizes, int n_in,
                              void* d_out, int out_size, void* d_ws, size_t ws_size,
                              hipStream_t stream) {
    const float* pred_logits = (const float*)d_in[0];  // [16,900,256]
    const float* pred_boxes  = (const float*)d_in[1];  // [16,900,4]
    const float* tgt_embs    = (const float*)d_in[2];  // [1600,256]
    const float* tgt_bbox    = (const float*)d_in[3];  // [1600,4]
    // d_in[4] = tgt_sizes: host-side LSA only, unused here.

    float* out = (float*)d_out;                        // [16,900,1600]

    _Float16* A = (_Float16*)d_ws;                     // NROWS*NDIM halves
    _Float16* B = A + (size_t)NROWS * NDIM;            // NTGT*NDIM halves

    // 1) normalize logits -> f16 A; convert embs -> f16 B
    matcher_prep_kernel<<<NROWS + NTGT, 256, 0, stream>>>(pred_logits, tgt_embs, A, B);

    // 2) fused WMMA cost volume
    dim3 grid(NROWS / 16, (NTGT / 16) / 4);            // (900, 25)
    matcher_cost_kernel<<<grid, 128, 0, stream>>>(A, B, pred_boxes, tgt_bbox, out);
}
